// MultiheadAttention_14104672600721
// MI455X (gfx1250) — matrix operand order repaired and checked by reading, not compiled, not run on hardware
//
#include <hip/hip_runtime.h>
#include <hip/hip_bf16.h>

// ---------------------------------------------------------------------------
// MultiheadAttention fwd for MI455X (gfx1250, wave32, WMMA).
//   B=2, S=2048, D=1024, H=16, HD=64.  ~69 GFLOP (~52 causal), working set < L2 (192MB)
//   -> compute bound -> v_wmma_f32_16x16x32_bf16 everywhere, f32 softmax.
// ---------------------------------------------------------------------------

#define BB 2
#define SS 2048
#define DM 1024
#define HH 16
#define HD 64
#define MM (BB * SS)   // 4096 rows in all GEMMs

typedef __attribute__((ext_vector_type(16))) __bf16 v16bf;
typedef __attribute__((ext_vector_type(8)))  float  v8f;

// ---- WMMA fragment helpers (ISA 7.12.2 layouts, wave32) -------------------
__device__ __forceinline__ int kmapA(int v, int half) {
    // 16-bit A 16x32: lanes0-15 K=0..7,16..23 ; lanes16-31 K=8..15,24..31
    return ((v >> 2) << 4) + (half << 3) + ((v & 3) << 1);
}
__device__ __forceinline__ int kmapB(int v, int half) {
    // 16-bit B 32x16: lanes0-15 K=0..7,16..23 ; lanes16-31 K=8..15,24..31
    return ((v >> 2) << 4) + (half << 3) + ((v & 3) << 1);
}

__device__ __forceinline__ v16bf load_fragA(const __bf16* base, int stride, int r, int half) {
    union { v16bf v; unsigned u[8]; } f;
#pragma unroll
    for (int v_ = 0; v_ < 8; ++v_)
        f.u[v_] = *(const unsigned*)(base + r * stride + kmapA(v_, half));
    return f.v;
}
__device__ __forceinline__ v16bf load_fragB(const __bf16* base, int stride, int r, int half) {
    union { v16bf v; unsigned u[8]; } f;
#pragma unroll
    for (int v_ = 0; v_ < 8; ++v_)
        f.u[v_] = *(const unsigned*)(base + r * stride + kmapB(v_, half));
    return f.v;
}
__device__ __forceinline__ v8f wmma_bf16(v16bf a, v16bf b, v8f c) {
    return __builtin_amdgcn_wmma_f32_16x16x32_bf16(false, a, false, b, (short)0, c, false, false);
}

// ---------------------------------------------------------------------------
// f32 -> bf16 conversion (x)
// ---------------------------------------------------------------------------
__global__ void __launch_bounds__(256) cvt_bf16_kernel(const float* __restrict__ src,
                                                       __bf16* __restrict__ dst, int n) {
    int i = blockIdx.x * 256 + threadIdx.x;
    if (i < n) dst[i] = (__bf16)src[i];
}

// ---------------------------------------------------------------------------
// f32 W[k][n] -> bf16 Wt[n][k], LDS-tiled transpose (coalesced both sides).
// Done ONCE per weight so the GEMM never transposes in its inner loop.
// ---------------------------------------------------------------------------
__global__ void __launch_bounds__(256) cvt_transpose_kernel(const float* __restrict__ src,
                                                            __bf16* __restrict__ dst) {
    __shared__ float t[32][33];
    const int bx = blockIdx.x * 32;   // n tile
    const int by = blockIdx.y * 32;   // k tile
    const int tx = threadIdx.x & 31;
    const int ty = threadIdx.x >> 5;  // 0..7
#pragma unroll
    for (int j = 0; j < 4; ++j)
        t[ty + 8 * j][tx] = src[(long)(by + ty + 8 * j) * DM + bx + tx];
    __syncthreads();
#pragma unroll
    for (int j = 0; j < 4; ++j)
        dst[(long)(bx + ty + 8 * j) * DM + by + tx] = (__bf16)t[tx][ty + 8 * j];
}

// ---------------------------------------------------------------------------
// Tiled WMMA GEMM:  C[M=4096, N=1024] = A[4096,1024] @ W[1024,1024] + bias
//   A: bf16 [M][K] row-major;  Wt: bf16 [N][K] (pre-transposed).
//   block = 256 thr (8 waves), block tile 64(M) x 64(N); each wave owns two
//   16x16 C tiles sharing one A fragment. K staged through LDS in chunks of
//   64 (4 WMMA / wave / step), register double-buffered global loads.
// mode: 0 = Q (RoPE, bf16 [B,H,S,HD])   1 = K (RoPE, bf16 [B,H,S,HD])
//       2 = V (bf16 transposed [B,H,HD,S])   3 = out-proj (f32 [B,S,D])
// ---------------------------------------------------------------------------
__global__ void __launch_bounds__(256) gemm_kernel(const __bf16* __restrict__ A,
                                                   const __bf16* __restrict__ Wt,
                                                   const float* __restrict__ bias,
                                                   __bf16* __restrict__ out_bf,
                                                   float* __restrict__ out_f,
                                                   int mode) {
    __shared__ __align__(16) __bf16 As[64][80];  // [m][k]  (64 used + pad)
    __shared__ __align__(16) __bf16 Bs[64][80];  // [n][k]

    const int tid  = threadIdx.x;
    const int wid  = tid >> 5;
    const int lane = tid & 31;
    const int r    = lane & 15;
    const int half = lane >> 4;
    const int m0   = blockIdx.x * 64;
    const int n0   = blockIdx.y * 64;
    const int wm   = (wid & 3) << 4;   // 0/16/32/48
    const int wn0  = (wid >> 2) << 5;  // 0/32  (two 16-wide n tiles)

    const int row = tid >> 2;          // 0..63
    const int kg  = (tid & 3) << 4;    // 0/16/32/48 (16 bf16 = 2 x b128)

    const __bf16* Ap = A  + (long)(m0 + row) * DM + kg;
    const __bf16* Bp = Wt + (long)(n0 + row) * DM + kg;

    uint4 ra0 = *(const uint4*)(Ap);
    uint4 ra1 = *(const uint4*)(Ap + 8);
    uint4 rb0 = *(const uint4*)(Bp);
    uint4 rb1 = *(const uint4*)(Bp + 8);

    v8f acc[2];
#pragma unroll
    for (int nn = 0; nn < 2; ++nn)
#pragma unroll
        for (int v_ = 0; v_ < 8; ++v_) acc[nn][v_] = 0.f;

    for (int k0 = 0; k0 < DM; k0 += 64) {
        __syncthreads();  // LDS free from previous compute
        *(uint4*)&As[row][kg]     = ra0;
        *(uint4*)&As[row][kg + 8] = ra1;
        *(uint4*)&Bs[row][kg]     = rb0;
        *(uint4*)&Bs[row][kg + 8] = rb1;
        __syncthreads();
        if (k0 + 64 < DM) {  // prefetch next tiles while WMMAs run
            ra0 = *(const uint4*)(Ap + k0 + 64);
            ra1 = *(const uint4*)(Ap + k0 + 72);
            rb0 = *(const uint4*)(Bp + k0 + 64);
            rb1 = *(const uint4*)(Bp + k0 + 72);
        }
#pragma unroll
        for (int c = 0; c < 2; ++c) {
            v16bf af  = load_fragA(&As[wm][c * 32], 80, r, half);
            v16bf bf0 = load_fragB(&Bs[wn0][c * 32], 80, r, half);
            v16bf bf1 = load_fragB(&Bs[wn0 + 16][c * 32], 80, r, half);
            acc[0] = wmma_bf16(af, bf0, acc[0]);
            acc[1] = wmma_bf16(af, bf1, acc[1]);
        }
    }

    // ---- epilogue: bias (+ fused RoPE for Q/K) + layout store -------------
#pragma unroll
    for (int nn = 0; nn < 2; ++nn) {
#pragma unroll
        for (int v_ = 0; v_ < 8; ++v_) {
            int mrow = m0 + wm + v_ + (half << 3);
            int ncol = n0 + wn0 + nn * 16 + r;
            float val = acc[nn][v_] + bias[ncol];
            if (mode <= 1) {  // RoPE pairs: adjacent columns = adjacent lanes
                int s = mrow & (SS - 1);
                int d = ncol & (HD - 1);
                float part  = __shfl_xor(val, 1, 32);
                float theta = __expf(-(float)(d & ~1) * 0.14391157f);  // ln(1e4)/64
                float sn, cs;
                __sincosf((float)s * theta, &sn, &cs);
                val = (d & 1) ? (val * cs + part * sn) : (val * cs - part * sn);
            }
            if (mode == 3) {
                out_f[(long)mrow * DM + ncol] = val;
            } else {
                int b = mrow >> 11, s = mrow & (SS - 1);
                int h = ncol >> 6,  d = ncol & (HD - 1);
                long idx = (mode == 2)
                               ? (((long)(b * HH + h)) * HD + d) * SS + s    // V^T
                               : (((long)(b * HH + h)) * SS + s) * HD + d;   // Q,K
                out_bf[idx] = (__bf16)val;
            }
        }
    }
}

// ---------------------------------------------------------------------------
// Causal flash attention. One wave <-> one 16-query tile, 32 keys per step.
// Q,K : bf16 [B,H,S,HD]   Vt : bf16 [B,H,HD,S]   out: bf16 [B,S,H*HD]
// ---------------------------------------------------------------------------
__global__ void __launch_bounds__(256) flash_kernel(const __bf16* __restrict__ Q,
                                                    const __bf16* __restrict__ K,
                                                    const __bf16* __restrict__ Vt,
                                                    __bf16* __restrict__ O) {
    __shared__ __align__(16) __bf16 Pls[8][16][36];  // per-wave P staging

    const int tid  = threadIdx.x;
    const int wid  = tid >> 5;
    const int lane = tid & 31;
    const int r    = lane & 15;
    const int half = lane >> 4;
    const int bh   = blockIdx.y;
    const int q0   = blockIdx.x * 128 + wid * 16;

    const __bf16* Qb = Q  + ((long)bh * SS + q0) * HD;
    const __bf16* Kb = K  + (long)bh * SS * HD;
    const __bf16* Vb = Vt + (long)bh * HD * SS;

    v16bf qf0 = load_fragA(Qb,      HD, r, half);
    v16bf qf1 = load_fragA(Qb + 32, HD, r, half);

    v8f o[4];
    float m8[8], l8[8];
#pragma unroll
    for (int j = 0; j < 4; ++j)
#pragma unroll
        for (int v_ = 0; v_ < 8; ++v_) o[j][v_] = 0.f;
#pragma unroll
    for (int v_ = 0; v_ < 8; ++v_) { m8[v_] = -3.0e38f; l8[v_] = 0.f; }

    const int kend = q0 + 16;  // exclusive causal bound
    for (int kk = 0; kk < kend; kk += 32) {
        if (kk + 32 < kend) {  // prefetch next chunk (global_prefetch_b8)
            __builtin_prefetch(Kb + (long)(kk + 32 + lane) * HD, 0, 0);
            __builtin_prefetch(Vb + (long)(lane * 2) * SS + kk + 32, 0, 0);
        }
        v8f st[2];
#pragma unroll
        for (int t = 0; t < 2; ++t) {
            int kt = kk + 16 * t;
            if (kt < kend) {
                v8f s = {0.f, 0.f, 0.f, 0.f, 0.f, 0.f, 0.f, 0.f};
                v16bf kf0 = load_fragB(Kb + (long)kt * HD,      HD, r, half);
                v16bf kf1 = load_fragB(Kb + (long)kt * HD + 32, HD, r, half);
                s = wmma_bf16(qf0, kf0, s);
                s = wmma_bf16(qf1, kf1, s);
#pragma unroll
                for (int v_ = 0; v_ < 8; ++v_) {
                    int row = q0 + v_ + (half << 3);
                    int key = kt + r;
                    float x = s[v_] * 0.125f;  // 1/sqrt(64)
                    st[t][v_] = (key <= row) ? x : -1.0e30f;
                }
            } else {
#pragma unroll
                for (int v_ = 0; v_ < 8; ++v_) st[t][v_] = -1.0e30f;
            }
        }
        // ---- online softmax; rows live across the 16-lane half ------------
        float alpha[8];
#pragma unroll
        for (int v_ = 0; v_ < 8; ++v_) {
            float mx = fmaxf(st[0][v_], st[1][v_]);
            mx = fmaxf(mx, __shfl_xor(mx, 1, 32));
            mx = fmaxf(mx, __shfl_xor(mx, 2, 32));
            mx = fmaxf(mx, __shfl_xor(mx, 4, 32));
            mx = fmaxf(mx, __shfl_xor(mx, 8, 32));
            float mn = fmaxf(m8[v_], mx);
            alpha[v_] = __expf(m8[v_] - mn);
            m8[v_] = mn;
            float p0 = __expf(st[0][v_] - mn);
            float p1 = __expf(st[1][v_] - mn);
            st[0][v_] = p0;
            st[1][v_] = p1;
            float rs = p0 + p1;
            rs += __shfl_xor(rs, 1, 32);
            rs += __shfl_xor(rs, 2, 32);
            rs += __shfl_xor(rs, 4, 32);
            rs += __shfl_xor(rs, 8, 32);
            l8[v_] = l8[v_] * alpha[v_] + rs;
        }
#pragma unroll
        for (int j = 0; j < 4; ++j)
#pragma unroll
            for (int v_ = 0; v_ < 8; ++v_) o[j][v_] *= alpha[v_];

        // ---- P: C-layout -> A-fragment via per-wave LDS tile --------------
#pragma unroll
        for (int t = 0; t < 2; ++t)
#pragma unroll
            for (int v_ = 0; v_ < 8; ++v_)
                Pls[wid][v_ + (half << 3)][t * 16 + r] = (__bf16)st[t][v_];
        asm volatile("s_wait_dscnt 0" ::: "memory");  // wave-local LDS sync
        v16bf pf = load_fragA(&Pls[wid][0][0], 36, r, half);

        // ---- O += P @ V  (V^T rows contiguous per lane) -------------------
#pragma unroll
        for (int j = 0; j < 4; ++j) {
            v16bf vf = load_fragB(Vb + (long)(16 * j) * SS + kk, SS, r, half);
            o[j] = wmma_bf16(pf, vf, o[j]);
        }
    }

    // ---- normalize + store merged-head layout [B,S,D] ---------------------
    const int b = bh >> 4, h = bh & 15;
#pragma unroll
    for (int j = 0; j < 4; ++j)
#pragma unroll
        for (int v_ = 0; v_ < 8; ++v_) {
            int srow = q0 + v_ + (half << 3);
            int d    = h * HD + j * 16 + r;
            O[((long)(b * SS + srow)) * DM + d] = (__bf16)(o[j][v_] / l8[v_]);
        }
}

// ---------------------------------------------------------------------------
// Host launcher
// ---------------------------------------------------------------------------
extern "C" void kernel_launch(void* const* d_in, const int* in_sizes, int n_in,
                              void* d_out, int out_size, void* d_ws, size_t ws_size,
                              hipStream_t stream) {
    (void)in_sizes; (void)n_in; (void)out_size; (void)ws_size;
    const float* x  = (const float*)d_in[0];
    // d_in[1] = mask (causal, implemented directly)
    const float* wq = (const float*)d_in[2];
    const float* bq = (const float*)d_in[3];
    const float* wk = (const float*)d_in[4];
    const float* bk = (const float*)d_in[5];
    const float* wv = (const float*)d_in[6];
    const float* bv = (const float*)d_in[7];
    const float* wo = (const float*)d_in[8];
    const float* bo = (const float*)d_in[9];
    float* out = (float*)d_out;

    char* ws = (char*)d_ws;
    __bf16* xb  = (__bf16*)(ws);                        // 8 MB  bf16 x
    __bf16* wqt = (__bf16*)(ws + ((size_t)8  << 20));   // 2 MB  bf16 Wq^T
    __bf16* wkt = (__bf16*)(ws + ((size_t)10 << 20));
    __bf16* wvt = (__bf16*)(ws + ((size_t)12 << 20));
    __bf16* wot = (__bf16*)(ws + ((size_t)14 << 20));
    __bf16* qb  = (__bf16*)(ws + ((size_t)16 << 20));   // 8 MB [B,H,S,HD]
    __bf16* kb  = (__bf16*)(ws + ((size_t)24 << 20));   // 8 MB [B,H,S,HD]
    __bf16* vt  = (__bf16*)(ws + ((size_t)32 << 20));   // 8 MB [B,H,HD,S]
    __bf16* at  = (__bf16*)(ws + ((size_t)40 << 20));   // 8 MB [B,S,D]

    const int nx = MM * DM;  // 4194304
    cvt_bf16_kernel<<<nx / 256, 256, 0, stream>>>(x, xb, nx);

    dim3 tgrid(DM / 32, DM / 32);  // (32, 32)
    cvt_transpose_kernel<<<tgrid, 256, 0, stream>>>(wq, wqt);
    cvt_transpose_kernel<<<tgrid, 256, 0, stream>>>(wk, wkt);
    cvt_transpose_kernel<<<tgrid, 256, 0, stream>>>(wv, wvt);
    cvt_transpose_kernel<<<tgrid, 256, 0, stream>>>(wo, wot);

    dim3 ggrid(MM / 64, DM / 64);  // (64, 16)
    gemm_kernel<<<ggrid, 256, 0, stream>>>(xb, wqt, bq, qb, nullptr, 0);
    gemm_kernel<<<ggrid, 256, 0, stream>>>(xb, wkt, bk, kb, nullptr, 1);
    gemm_kernel<<<ggrid, 256, 0, stream>>>(xb, wvt, bv, vt, nullptr, 2);

    dim3 fgrid(SS / 128, BB * HH);  // (16, 32)
    flash_kernel<<<fgrid, 256, 0, stream>>>(qb, kb, vt, at);

    gemm_kernel<<<ggrid, 256, 0, stream>>>(at, wot, bo, nullptr, out, 3);
}
